// SparseMultiheadAttention_31997506355767
// MI455X (gfx1250) — matrix operand order repaired and checked by reading, not compiled, not run on hardware
//
#include <hip/hip_runtime.h>

#define DMODEL 512
#define NTOK   8192   // B*L
#define LSEQ   1024
#define NHEADS 8
#define DHEAD  64
#define WPAD   264    // padded LDS row stride (bf16 elems) for the 256-wide W panel

typedef __bf16 bf16_t;
typedef __attribute__((ext_vector_type(16))) bf16_t v16bf;
typedef __attribute__((ext_vector_type(8)))  bf16_t v8bf;
typedef __attribute__((ext_vector_type(8)))  float  v8f;
typedef __attribute__((ext_vector_type(4)))  float  v4f;

#define WMMA_BF16(A, B, C) \
  __builtin_amdgcn_wmma_f32_16x16x32_bf16(false, (A), false, (B), (short)0, (C), false, false)

// ---- WMMA fragment loaders (CDNA5 wave32 layouts, CDNA5 ISA §7.12.2) ----
// A (16x32, 16-bit): lane = row (mod 16); k order used here (B matches): element e -> K = e + 16*(lane/16)
static __device__ __forceinline__ v16bf wmma_a_from_bf16(const bf16_t* row, int k0, int half) {
  const bf16_t* p = row + k0 + 16 * half;
  v8bf g0 = *(const v8bf*)(p);
  v8bf g1 = *(const v8bf*)(p + 8);
  v16bf r;
#pragma unroll
  for (int i = 0; i < 8; ++i) { r[i] = g0[i]; r[i + 8] = g1[i]; }
  return r;
}

// B fragment from 16 contiguous bf16 (16B-aligned is enough: two v8bf loads)
static __device__ __forceinline__ v16bf wmma_b_contig(const bf16_t* p) {
  v8bf g0 = *(const v8bf*)(p);
  v8bf g1 = *(const v8bf*)(p + 8);
  v16bf r;
#pragma unroll
  for (int i = 0; i < 8; ++i) { r[i] = g0[i]; r[i + 8] = g1[i]; }
  return r;
}

// A fragment sourced from f32 (LDS scores) with conversion
static __device__ __forceinline__ v16bf wmma_a_from_f32(const float* row, int k0, int half) {
  const float* p = row + k0 + 16 * half;
  v4f g0 = *(const v4f*)(p);
  v4f g1 = *(const v4f*)(p + 4);
  v4f g2 = *(const v4f*)(p + 8);
  v4f g3 = *(const v4f*)(p + 12);
  v16bf r;
#pragma unroll
  for (int i = 0; i < 4; ++i) {
    r[i]      = (bf16_t)g0[i];
    r[i + 4]  = (bf16_t)g1[i];
    r[i + 8]  = (bf16_t)g2[i];
    r[i + 12] = (bf16_t)g3[i];
  }
  return r;
}

// ---------------- Kernel 0: elementwise f32 -> bf16 conversion ----------------
__global__ __launch_bounds__(256)
void cvt_bf16_kernel(const float* __restrict__ src, bf16_t* __restrict__ dst, int n)
{
  const int i = (blockIdx.x * 256 + threadIdx.x) * 8;
  if (i + 8 <= n) {
    v8f v = *(const v8f*)(src + i);
    v8bf o;
#pragma unroll
    for (int j = 0; j < 8; ++j) o[j] = (bf16_t)v[j];
    *(v8bf*)(dst + i) = o;
  }
}

// ---------------- Kernel 1: GEMM  Y = A @ W^T + b   (all-bf16, W staged in LDS) -------
// M=8192, N=512, K=512. 8 waves/block; wave -> 32x64 tile (2 m-tiles x 4 n-tiles).
// The block's 64-row W tile is staged in LDS in two K=256 panels, so each weight byte
// is fetched from global exactly once per block (8x traffic cut vs. per-wave loads).
// mode 0: bf16 row-major store; mode 1: V head-transposed store; mode 2: f32 row store
__global__ __launch_bounds__(256)
void gemm_bf16_kernel(const bf16_t* __restrict__ A, const bf16_t* __restrict__ W,
                      const float* __restrict__ bias,
                      bf16_t* __restrict__ Yrow, bf16_t* __restrict__ Ytr,
                      float* __restrict__ Yf32, int mode)
{
  __shared__ bf16_t sW[64 * WPAD];   // 33 KB: 64 n-rows x 256 k (padded)

  const int lane = threadIdx.x & 31;
  const int wave = threadIdx.x >> 5;
  const int col  = lane & 15;
  const int half = lane >> 4;
  const int m0 = blockIdx.y * 256 + wave * 32;
  const int n0 = blockIdx.x * 64;

  const bf16_t* arow0 = A + (size_t)(m0 + col) * DMODEL;
  const bf16_t* arow1 = arow0 + (size_t)16 * DMODEL;

  v8f acc[2][4] = {};

  // A pipeline primed for k=0
  v16bf a0n = wmma_a_from_bf16(arow0, 0, half);
  v16bf a1n = wmma_a_from_bf16(arow1, 0, half);

  for (int phase = 0; phase < 2; ++phase) {
    const int kbase = phase * 256;

    // ---- cooperative stage: 256 threads copy 64x256 bf16 of W into LDS ----
    {
      const int r = threadIdx.x >> 2;        // 0..63  (n-row)
      const int q = threadIdx.x & 3;
      const bf16_t* gsrc = W + (size_t)(n0 + r) * DMODEL + kbase;
      bf16_t*       ldst = sW + r * WPAD;
#pragma unroll
      for (int i = 0; i < 8; ++i) {
        const int c = (q + 4 * i) * 8;       // 8-elem (16B) chunks
        *(v8bf*)(ldst + c) = *(const v8bf*)(gsrc + c);
      }
    }
    __syncthreads();

    // ---- inner k loop over this panel; B from LDS, A pipelined from global ----
    const bf16_t* swb = sW + col * WPAD + 16 * half;
    for (int kk = 0; kk < 256; kk += 32) {
      const v16bf a0 = a0n, a1 = a1n;
      const int knext = kbase + kk + 32;
      if (knext < DMODEL) {                  // prefetch next A fragments
        a0n = wmma_a_from_bf16(arow0, knext, half);
        a1n = wmma_a_from_bf16(arow1, knext, half);
      }
      const v16bf b0 = wmma_b_contig(swb + 0 * 16 * WPAD + kk);
      const v16bf b1 = wmma_b_contig(swb + 1 * 16 * WPAD + kk);
      const v16bf b2 = wmma_b_contig(swb + 2 * 16 * WPAD + kk);
      const v16bf b3 = wmma_b_contig(swb + 3 * 16 * WPAD + kk);
      acc[0][0] = WMMA_BF16(a0, b0, acc[0][0]);
      acc[0][1] = WMMA_BF16(a0, b1, acc[0][1]);
      acc[0][2] = WMMA_BF16(a0, b2, acc[0][2]);
      acc[0][3] = WMMA_BF16(a0, b3, acc[0][3]);
      acc[1][0] = WMMA_BF16(a1, b0, acc[1][0]);
      acc[1][1] = WMMA_BF16(a1, b1, acc[1][1]);
      acc[1][2] = WMMA_BF16(a1, b2, acc[1][2]);
      acc[1][3] = WMMA_BF16(a1, b3, acc[1][3]);
    }
    __syncthreads();                         // LDS reuse across phases
  }

  // C/D layout: lane col = N, VGPR i -> row M = i + 8*(lane/16)
#pragma unroll
  for (int mt = 0; mt < 2; ++mt) {
#pragma unroll
    for (int nt = 0; nt < 4; ++nt) {
      const int n = n0 + nt * 16 + col;
      const float bv = bias[n];
#pragma unroll
      for (int i = 0; i < 8; ++i) {
        const int m = m0 + mt * 16 + i + 8 * half;
        const float v = acc[mt][nt][i] + bv;
        if (mode == 0) {
          Yrow[(size_t)m * DMODEL + n] = (bf16_t)v;
        } else if (mode == 1) {
          const int bb = m >> 10, j = m & 1023;
          const int hh = n >> 6,  d = n & 63;
          Ytr[(((size_t)(bb * NHEADS + hh) * DHEAD + d) << 10) + j] = (bf16_t)v;
        } else {
          Yf32[(size_t)m * DMODEL + n] = v;
        }
      }
    }
  }
}

// -------- Kernel 2: scores (WMMA) -> entmax-1.5 (bisection) -> ctx (WMMA) --------
// One wave per workgroup handles 16 query rows of one (b,h).  LDS: 16x1024 f32 scores.
__global__ __launch_bounds__(32)
void attn_entmax_kernel(const bf16_t* __restrict__ Qb, const bf16_t* __restrict__ Kb,
                        const bf16_t* __restrict__ Vt, const unsigned char* __restrict__ mask,
                        bf16_t* __restrict__ Ctx)
{
  __shared__ float sS[16 * LSEQ];   // 64 KB
  const int lane = threadIdx.x & 31;
  const int col  = lane & 15;
  const int half = lane >> 4;
  const int b  = blockIdx.z;
  const int h  = blockIdx.y;
  const int q0 = blockIdx.x * 16;

  // Q fragments (reused across all 64 j-tiles): 16 rows x 64 d, two K=32 chunks
  const bf16_t* qrow = Qb + ((size_t)b * LSEQ + q0 + col) * DMODEL + h * DHEAD;
  const v16bf aq0 = wmma_a_from_bf16(qrow, 0,  half);
  const v16bf aq1 = wmma_a_from_bf16(qrow, 32, half);

  // ---- scores S = (Q K^T) * d^-0.5, masked, into LDS (pipelined over j-tiles) ----
  const bf16_t* kbase = Kb + ((size_t)b * LSEQ + col) * DMODEL + h * DHEAD + 16 * half;
  v16bf bk0n = *(const v16bf*)(kbase);
  v16bf bk1n = *(const v16bf*)(kbase + 32);
  for (int jt = 0; jt < LSEQ / 16; ++jt) {
    const v16bf bk0 = bk0n, bk1 = bk1n;
    if (jt + 1 < LSEQ / 16) {
      const bf16_t* krow = kbase + (size_t)(jt + 1) * 16 * DMODEL;
      bk0n = *(const v16bf*)(krow);
      bk1n = *(const v16bf*)(krow + 32);
    }
    v8f acc = {};
    acc = WMMA_BF16(aq0, bk0, acc);
    acc = WMMA_BF16(aq1, bk1, acc);
    const int j = jt * 16 + col;
    const bool mm = mask[(size_t)b * LSEQ + j] != 0;
#pragma unroll
    for (int i = 0; i < 8; ++i) {
      const float v = acc[i] * 0.125f;              // D_HEAD^-0.5
      sS[(i + 8 * half) * LSEQ + j] = mm ? -1.0e9f : v;
    }
  }
  __syncthreads();

  // ---- entmax-1.5 per row: x=(s-max)/2, bisect tau in [-1,0] s.t. sum(relu(x-tau)^2)=1
  for (int r = 0; r < 16; ++r) {
    float x[32];
    float mx = -3.0e38f;
#pragma unroll
    for (int i = 0; i < 32; ++i) {
      const float v = sS[r * LSEQ + lane + 32 * i];
      x[i] = v;
      mx = fmaxf(mx, v);
    }
#pragma unroll
    for (int off = 16; off >= 1; off >>= 1) mx = fmaxf(mx, __shfl_xor(mx, off, 32));
#pragma unroll
    for (int i = 0; i < 32; ++i) x[i] = (x[i] - mx) * 0.5f;

    float lo = -1.0f, hi = 0.0f;
    for (int it = 0; it < 30; ++it) {
      const float mid = 0.5f * (lo + hi);
      float s = 0.0f;
#pragma unroll
      for (int i = 0; i < 32; ++i) {
        float t = x[i] - mid;
        t = t > 0.0f ? t : 0.0f;
        s += t * t;
      }
#pragma unroll
      for (int off = 16; off >= 1; off >>= 1) s += __shfl_xor(s, off, 32);
      if (s >= 1.0f) lo = mid; else hi = mid;
    }
    const float tau = 0.5f * (lo + hi);
#pragma unroll
    for (int i = 0; i < 32; ++i) {
      float t = x[i] - tau;
      t = t > 0.0f ? t : 0.0f;
      sS[r * LSEQ + lane + 32 * i] = t * t;        // P overwrites S in LDS
    }
  }
  __syncthreads();

  // ---- ctx = P (16x1024) @ V (1024x64), V head-transposed, pipelined k loop ----
  v8f acc[4] = {};
  const float*  prow = sS + col * LSEQ;
  const bf16_t* vb   = Vt + ((size_t)(b * NHEADS + h) * DHEAD + col) * LSEQ + 16 * half;

  v16bf an  = wmma_a_from_f32(prow, 0, half);
  v16bf b0n = *(const v16bf*)(vb + 0 * 16 * LSEQ);
  v16bf b1n = *(const v16bf*)(vb + 1 * 16 * LSEQ);
  v16bf b2n = *(const v16bf*)(vb + 2 * 16 * LSEQ);
  v16bf b3n = *(const v16bf*)(vb + 3 * 16 * LSEQ);
  for (int k0 = 0; k0 < LSEQ; k0 += 32) {
    const v16bf a = an, b0 = b0n, b1 = b1n, b2 = b2n, b3 = b3n;
    const int kn = k0 + 32;
    if (kn < LSEQ) {
      an  = wmma_a_from_f32(prow, kn, half);
      b0n = *(const v16bf*)(vb + 0 * 16 * LSEQ + kn);
      b1n = *(const v16bf*)(vb + 1 * 16 * LSEQ + kn);
      b2n = *(const v16bf*)(vb + 2 * 16 * LSEQ + kn);
      b3n = *(const v16bf*)(vb + 3 * 16 * LSEQ + kn);
    }
    acc[0] = WMMA_BF16(a, b0, acc[0]);
    acc[1] = WMMA_BF16(a, b1, acc[1]);
    acc[2] = WMMA_BF16(a, b2, acc[2]);
    acc[3] = WMMA_BF16(a, b3, acc[3]);
  }
#pragma unroll
  for (int nt = 0; nt < 4; ++nt) {
#pragma unroll
    for (int i = 0; i < 8; ++i) {
      Ctx[((size_t)b * LSEQ + q0 + i + 8 * half) * DMODEL + h * DHEAD + nt * 16 + col] =
          (bf16_t)acc[nt][i];
    }
  }
}

extern "C" void kernel_launch(void* const* d_in, const int* in_sizes, int n_in,
                              void* d_out, int out_size, void* d_ws, size_t ws_size,
                              hipStream_t stream)
{
  const float* query = (const float*)d_in[0];
  const float* key_  = (const float*)d_in[1];
  const float* value = (const float*)d_in[2];
  const unsigned char* mask = (const unsigned char*)d_in[3];
  const float* q_w   = (const float*)d_in[4];
  const float* q_b   = (const float*)d_in[5];
  const float* k_w   = (const float*)d_in[6];
  const float* k_b   = (const float*)d_in[7];
  const float* v_w   = (const float*)d_in[8];
  const float* v_b   = (const float*)d_in[9];
  const float* out_w = (const float*)d_in[10];
  const float* out_b = (const float*)d_in[11];

  const size_t elems  = (size_t)NTOK * DMODEL;    // 4,194,304
  const size_t welems = (size_t)DMODEL * DMODEL;  // 262,144
  const size_t need   = (5 * elems + 4 * welems) * sizeof(bf16_t);  // ~42 MB
  if (ws_size < need) return;

  bf16_t* buf0 = (bf16_t*)d_ws;        // Xq, later Ctx
  bf16_t* buf1 = buf0 + elems;         // Xk, later Vt
  bf16_t* buf2 = buf1 + elems;         // Xv
  bf16_t* Qb   = buf2 + elems;
  bf16_t* Kb   = Qb + elems;
  bf16_t* Wq   = Kb + elems;
  bf16_t* Wk   = Wq + welems;
  bf16_t* Wv   = Wk + welems;
  bf16_t* Wo   = Wv + welems;

  // 0) convert activations + weights to bf16
  const int nblkA = (int)(elems / 8 / 256);
  const int nblkW = (int)(welems / 8 / 256);
  cvt_bf16_kernel<<<nblkA, 256, 0, stream>>>(query, buf0, (int)elems);
  cvt_bf16_kernel<<<nblkA, 256, 0, stream>>>(key_,  buf1, (int)elems);
  cvt_bf16_kernel<<<nblkA, 256, 0, stream>>>(value, buf2, (int)elems);
  cvt_bf16_kernel<<<nblkW, 256, 0, stream>>>(q_w,   Wq,   (int)welems);
  cvt_bf16_kernel<<<nblkW, 256, 0, stream>>>(k_w,   Wk,   (int)welems);
  cvt_bf16_kernel<<<nblkW, 256, 0, stream>>>(v_w,   Wv,   (int)welems);
  cvt_bf16_kernel<<<nblkW, 256, 0, stream>>>(out_w, Wo,   (int)welems);

  // 1) projections  (V stored head-transposed into buf1; Xk is dead by then)
  const dim3 gG(DMODEL / 64, NTOK / 256);
  gemm_bf16_kernel<<<gG, 256, 0, stream>>>(buf0, Wq, q_b, Qb, nullptr, nullptr, 0);
  gemm_bf16_kernel<<<gG, 256, 0, stream>>>(buf1, Wk, k_b, Kb, nullptr, nullptr, 0);
  gemm_bf16_kernel<<<gG, 256, 0, stream>>>(buf2, Wv, v_b, nullptr, buf1, nullptr, 1);

  // 2) attention + entmax-1.5  (Ctx into buf0; Xq is dead)
  attn_entmax_kernel<<<dim3(LSEQ / 16, NHEADS, 8), 32, 0, stream>>>(Qb, Kb, buf1, mask, buf0);

  // 3) output projection -> fp32 d_out
  gemm_bf16_kernel<<<gG, 256, 0, stream>>>(buf0, Wo, out_b, nullptr, nullptr, (float*)d_out, 2);
}
